// SSA_47382079210109
// MI455X (gfx1250) — compile-verified, matrix operand order checked by reading, not run on hardware
//
#include <hip/hip_runtime.h>

typedef __attribute__((ext_vector_type(16))) _Float16 v16h;
typedef __attribute__((ext_vector_type(8)))  float    v8f;
typedef __attribute__((ext_vector_type(4)))  float    v4f;
typedef unsigned int u32;

#define HID   384
#define NTOK  2048     // 16*16*8
#define BATCH 2
#define HEADS 8
#define DH    48
#define QKVC  1152
#define QK_SCALE 0.14433756729740643f   // 48^-0.5

union U32H { u32 u; _Float16 h[2]; };

__device__ __forceinline__ v8f wmma16(v16h a, v16h b, v8f c) {
  // D = A(16x32 f16) * B(32x16 f16) + C(16x16 f32)
  return __builtin_amdgcn_wmma_f32_16x16x32_f16(false, a, false, b, (short)0, c,
                                                false, false);
}

// A/B fragment gather from a row-major tile row (>=32 valid halves).
// Layout: K = h + 8*(h>=8) + 8*(lane>=16), per ISA 7.12.2 (16-bit 16x32).
__device__ __forceinline__ v16h frag_ld(const _Float16* rowbase, int lane) {
  const int off = (lane & 16) ? 8 : 0;
  v16h f;
#pragma unroll
  for (int h = 0; h < 8; ++h) f[h] = rowbase[off + h];
#pragma unroll
  for (int h = 0; h < 8; ++h) f[8 + h] = rowbase[off + 16 + h];
  return f;
}

// Fragment covering K-range [32..63] of a dh=48 row: halves 8..15 (dh 48..63)
// are zero so no padded storage is needed.
__device__ __forceinline__ v16h frag_ld_hi48(const _Float16* rowbase, int lane) {
  const int off = 32 + ((lane & 16) ? 8 : 0);
  v16h f;
#pragma unroll
  for (int h = 0; h < 8; ++h) f[h] = rowbase[off + h];
#pragma unroll
  for (int h = 0; h < 8; ++h) f[8 + h] = (_Float16)0.0f;
  return f;
}

// Async global->LDS copy of 16B (ASYNCcnt-tracked, no VGPR data path).
__device__ __forceinline__ void async_cp16(u32 lds_off, const void* gptr) {
  asm volatile("global_load_async_to_lds_b128 %0, %1, off"
               :: "v"(lds_off), "v"((unsigned long long)(uintptr_t)gptr)
               : "memory");
}
__device__ __forceinline__ void wait_async0() {
  asm volatile("s_wait_asynccnt 0x0" ::: "memory");
}
__device__ __forceinline__ u32 lds_off32(const void* p) {
  return (u32)(uintptr_t)p;  // generic LDS addr: low 32 bits = LDS byte offset
}

// ---------------------------------------------------------------- kernel 1
// tokens[m=(b,n), c] = x[b,c,n];  qkv = tokens @ w_qkv  -> q,k,v f16 buffers
__global__ __launch_bounds__(256) void qkv_kernel(
    const float* __restrict__ x, const float* __restrict__ w_qkv,
    _Float16* __restrict__ qb, _Float16* __restrict__ kb,
    _Float16* __restrict__ vb) {
  __shared__ __align__(16) _Float16 At[128 * 32];
  __shared__ __align__(16) _Float16 Bt[64 * 32];
  const int t = threadIdx.x;
  const int lane = t & 31;
  const int wv = t >> 5;
  const int colBase = blockIdx.x * 64;   // 18 blocks over 1152
  const int rowBase = blockIdx.y * 128;  // 32 blocks over 4096

  v8f acc[4] = {};

  for (int kk = 0; kk < 12; ++kk) {
    // ---- A tile 128x32: batched float4 loads (coalesced over n), then cvt
    v4f atmp[4];
#pragma unroll
    for (int rep = 0; rep < 4; ++rep) {
      int lin = rep * 256 + t;             // 1024 float4 groups
      int k = lin >> 5, i4 = lin & 31;
      int m = rowBase + i4 * 4;
      int b = m >> 11, n = m & 2047;
      atmp[rep] = *(const v4f*)&x[((size_t)b * HID + kk * 32 + k) * NTOK + n];
    }
    // ---- B tile: [k][j] -> LDS [j][k], batched float4 loads
    v4f btmp[2];
#pragma unroll
    for (int rep = 0; rep < 2; ++rep) {
      int lin = rep * 256 + t;             // 512 float4 groups
      int k = lin >> 4, j4 = lin & 15;
      btmp[rep] =
          *(const v4f*)&w_qkv[(size_t)(kk * 32 + k) * QKVC + colBase + j4 * 4];
    }
#pragma unroll
    for (int rep = 0; rep < 4; ++rep) {
      int lin = rep * 256 + t;
      int k = lin >> 5, i4 = lin & 31;
#pragma unroll
      for (int j = 0; j < 4; ++j)
        At[(i4 * 4 + j) * 32 + k] = (_Float16)atmp[rep][j];
    }
#pragma unroll
    for (int rep = 0; rep < 2; ++rep) {
      int lin = rep * 256 + t;
      int k = lin >> 4, j4 = lin & 15;
#pragma unroll
      for (int j = 0; j < 4; ++j)
        Bt[(j4 * 4 + j) * 32 + k] = (_Float16)btmp[rep][j];
    }
    __syncthreads();
    v16h a = frag_ld(&At[(wv * 16 + (lane & 15)) * 32], lane);
#pragma unroll
    for (int jt = 0; jt < 4; ++jt) {
      v16h bf = frag_ld(&Bt[(jt * 16 + (lane & 15)) * 32], lane);
      acc[jt] = wmma16(a, bf, acc[jt]);
    }
    __syncthreads();
  }

  // Branchless epilogue: 'which' is block-uniform (64 | colBase, 384 % 64 == 0)
  const int which = colBase / 384;
  _Float16* dstbuf = (which == 0) ? qb : ((which == 1) ? kb : vb);
  const float lo   = (which == 2) ? -3.0e38f : -5.0f;
  const float hi   = (which == 2) ?  3.0e38f :  5.0f;
  const float mulf = (which == 0) ? QK_SCALE : 1.0f;
  const int innerBase = colBase - which * 384;
  const int mrow = (lane & 16) ? 8 : 0;
#pragma unroll
  for (int jt = 0; jt < 4; ++jt) {
    int inner0 = innerBase + jt * 16;
    int head = inner0 / 48;              // uniform per jt (48 % 16 == 0)
    int dh = (inner0 % 48) + (lane & 15);
#pragma unroll
    for (int r = 0; r < 8; ++r) {
      int m = rowBase + wv * 16 + mrow + r;
      int b = m >> 11, n = m & 2047;
      float v = fminf(fmaxf(acc[jt][r], lo), hi) * mulf;
      dstbuf[((size_t)(b * HEADS + head) * NTOK + n) * DH + dh] = (_Float16)v;
    }
  }
}

// ---------------------------------------------------------------- kernel 2
// flash attention: block = (b,h) x 128 queries; wave = 16 query rows.
// Software-pipelined: async K tile + register-held V tile prefetch for step
// i+1 overlap the WMMA/softmax compute of step i; one barrier per step.
__global__ __launch_bounds__(256) void attn_kernel(
    const _Float16* __restrict__ qb, const _Float16* __restrict__ kb,
    const _Float16* __restrict__ vb, _Float16* __restrict__ ao) {
  __shared__ __align__(16) _Float16 Kl[2][32 * 48];  // [buf][kv][dh]
  __shared__ __align__(16) _Float16 Vt[2][48 * 32];  // [buf][dh][kv]
  __shared__ __align__(16) _Float16 Pb[8][16 * 32];
  const int t = threadIdx.x;
  const int lane = t & 31;
  const int wv = t >> 5;
  const int qBase = blockIdx.x * 128;  // 16 blocks over 2048
  const int bh = blockIdx.y;           // b*8 + head
  const int mrow = (lane & 16) ? 8 : 0;

  const int qrow = qBase + wv * 16 + (lane & 15);
  const _Float16* qp = qb + ((size_t)bh * NTOK + qrow) * DH;
  v16h qa0 = frag_ld(qp, lane);        // dh 0..31
  v16h qa1 = frag_ld_hi48(qp, lane);   // dh 32..47 (+zero pad)

  float rm[8], rl[8];
  v8f acc[3] = {};
#pragma unroll
  for (int r = 0; r < 8; ++r) { rm[r] = -3.0e38f; rl[r] = 0.f; }

  const u32 kl_base = lds_off32(Kl);
  const char* kgbase = (const char*)(kb + (size_t)bh * NTOK * DH);
  const u32* vgbase = (const u32*)(vb + (size_t)bh * NTOK * DH);

  // ---- preload tile 0
  if (t < 192) async_cp16(kl_base + (u32)(t * 16), kgbase + t * 16);
  {
    u32 vtmp[3];
#pragma unroll
    for (int rep = 0; rep < 3; ++rep) vtmp[rep] = vgbase[rep * 256 + t];
#pragma unroll
    for (int rep = 0; rep < 3; ++rep) {
      int lh = (rep * 256 + t) * 2;
      int kv = lh / 48, dh = lh % 48;
      U32H cv; cv.u = vtmp[rep];
      Vt[0][dh * 32 + kv] = cv.h[0];
      Vt[0][(dh + 1) * 32 + kv] = cv.h[1];
    }
  }
  wait_async0();
  __syncthreads();

  for (int it = 0; it < NTOK / 32; ++it) {
    const int cur = it & 1, nxt = cur ^ 1;
    const bool more = (it + 1) < NTOK / 32;

    // ---- prefetch tile it+1 (async K -> LDS[nxt], V -> registers)
    u32 vtmpN[3];
    if (more) {
      const size_t kvoff = (size_t)(it + 1) * 32 * DH;
      if (t < 192)
        async_cp16(kl_base + (u32)(nxt * 3072 + t * 16),
                   kgbase + kvoff * 2 + t * 16);
#pragma unroll
      for (int rep = 0; rep < 3; ++rep)
        vtmpN[rep] = vgbase[kvoff / 2 + rep * 256 + t];
    }

    // ---- compute on tile it
    const _Float16* Klc = &Kl[cur][0];
    const _Float16* Vtc = &Vt[cur][0];
    v8f s[2];
#pragma unroll
    for (int tt = 0; tt < 2; ++tt) {   // S = Q K^T (scale folded into q)
      const _Float16* krow = &Klc[(tt * 16 + (lane & 15)) * 48];
      v16h kf0 = frag_ld(krow, lane);
      v16h kf1 = frag_ld_hi48(krow, lane);
      v8f z = {};
      s[tt] = wmma16(qa0, kf0, z);
      s[tt] = wmma16(qa1, kf1, s[tt]);
    }

#pragma unroll
    for (int r = 0; r < 8; ++r) {      // online softmax update
      float ms = fmaxf(s[0][r], s[1][r]);
#pragma unroll
      for (int d = 1; d < 16; d <<= 1) ms = fmaxf(ms, __shfl_xor(ms, d, 32));
      float mn = fmaxf(rm[r], ms);
      float alpha = __expf(rm[r] - mn);
      rm[r] = mn;
      float p0 = __expf(s[0][r] - mn);
      float p1 = __expf(s[1][r] - mn);
      s[0][r] = p0; s[1][r] = p1;
      float ls = p0 + p1;
#pragma unroll
      for (int d = 1; d < 16; d <<= 1) ls += __shfl_xor(ls, d, 32);
      rl[r] = rl[r] * alpha + ls;
      acc[0][r] *= alpha; acc[1][r] *= alpha; acc[2][r] *= alpha;
    }

    // D-layout -> A-layout via per-wave LDS round trip
    _Float16* pb = &Pb[wv][0];
#pragma unroll
    for (int r = 0; r < 8; ++r) {
      pb[(mrow + r) * 32 + (lane & 15)]      = (_Float16)s[0][r];
      pb[(mrow + r) * 32 + (lane & 15) + 16] = (_Float16)s[1][r];
    }
    asm volatile("s_wait_dscnt 0x0" ::: "memory");  // wave-local RAW on LDS
    v16h pa = frag_ld(&pb[(lane & 15) * 32], lane);
#pragma unroll
    for (int d = 0; d < 3; ++d) {      // O += P V
      v16h vf = frag_ld(&Vtc[(d * 16 + (lane & 15)) * 32], lane);
      acc[d] = wmma16(pa, vf, acc[d]);
    }

    // ---- commit prefetched V into LDS[nxt] (last read at it-1, fenced by
    // the end-of-iteration barrier), then close the pipeline stage.
    if (more) {
#pragma unroll
      for (int rep = 0; rep < 3; ++rep) {
        int lh = (rep * 256 + t) * 2;
        int kv = lh / 48, dh = lh % 48;
        U32H cv; cv.u = vtmpN[rep];
        Vt[nxt][dh * 32 + kv] = cv.h[0];
        Vt[nxt][(dh + 1) * 32 + kv] = cv.h[1];
      }
    }
    wait_async0();      // own async K writes (tile it+1) done before barrier
    __syncthreads();    // visibility of Kl[nxt]/Vt[nxt]; closes reads of cur
  }

  const int b = bh >> 3, head = bh & 7;
#pragma unroll
  for (int r = 0; r < 8; ++r) {
    int q = qBase + wv * 16 + mrow + r;
    float inv = 1.0f / rl[r];
#pragma unroll
    for (int d = 0; d < 3; ++d) {
      int col = head * DH + d * 16 + (lane & 15);
      ao[((size_t)b * NTOK + q) * HID + col] = (_Float16)(acc[d][r] * inv);
    }
  }
}

// ---------------------------------------------------------------- kernel 3
// out = x + (attn_out @ w_out + b_out), transposed back to [B,C,N]
__global__ __launch_bounds__(256) void proj_kernel(
    const _Float16* __restrict__ ao, const float* __restrict__ w_out,
    const float* __restrict__ b_out, const float* __restrict__ x,
    float* __restrict__ out) {
  __shared__ __align__(16) _Float16 At[128 * 32];
  __shared__ __align__(16) _Float16 Bt[64 * 32];
  const int t = threadIdx.x;
  const int lane = t & 31;
  const int wv = t >> 5;
  const int colBase = blockIdx.x * 64;   // 6 blocks over 384
  const int rowBase = blockIdx.y * 128;  // 32 blocks over 4096

  v8f acc[4] = {};
  const u32 at_base = lds_off32(At);

  for (int kk = 0; kk < 12; ++kk) {
    // A tile (already f16, straight copy): async global->LDS b128 x2
#pragma unroll
    for (int rep = 0; rep < 2; ++rep) {
      int lin = rep * 256 + t;             // 512 x 16B chunks
      int i = lin >> 2, c4 = lin & 3;
      const char* g = (const char*)ao +
                      ((size_t)(rowBase + i) * HID + kk * 32) * 2 + c4 * 16;
      async_cp16(at_base + (u32)(i * 64 + c4 * 16), g);
    }
    // B tile: [k][c] -> LDS [c][k], batched float4 loads
    v4f btmp[2];
#pragma unroll
    for (int rep = 0; rep < 2; ++rep) {
      int lin = rep * 256 + t;
      int k = lin >> 4, j4 = lin & 15;
      btmp[rep] =
          *(const v4f*)&w_out[(size_t)(kk * 32 + k) * HID + colBase + j4 * 4];
    }
#pragma unroll
    for (int rep = 0; rep < 2; ++rep) {
      int lin = rep * 256 + t;
      int k = lin >> 4, j4 = lin & 15;
#pragma unroll
      for (int j = 0; j < 4; ++j)
        Bt[(j4 * 4 + j) * 32 + k] = (_Float16)btmp[rep][j];
    }
    wait_async0();
    __syncthreads();
    v16h a = frag_ld(&At[(wv * 16 + (lane & 15)) * 32], lane);
#pragma unroll
    for (int jt = 0; jt < 4; ++jt) {
      v16h bf = frag_ld(&Bt[(jt * 16 + (lane & 15)) * 32], lane);
      acc[jt] = wmma16(a, bf, acc[jt]);
    }
    __syncthreads();
  }

  const int mrow = (lane & 16) ? 8 : 0;
#pragma unroll
  for (int jt = 0; jt < 4; ++jt) {
    int c = colBase + jt * 16 + (lane & 15);
    float bias = b_out[c];
#pragma unroll
    for (int r = 0; r < 8; ++r) {
      int m = rowBase + wv * 16 + mrow + r;
      int b = m >> 11, n = m & 2047;
      size_t idx = ((size_t)b * HID + c) * NTOK + n;
      out[idx] = x[idx] + acc[jt][r] + bias;
    }
  }
}

extern "C" void kernel_launch(void* const* d_in, const int* in_sizes, int n_in,
                              void* d_out, int out_size, void* d_ws,
                              size_t ws_size, hipStream_t stream) {
  const float* x     = (const float*)d_in[0];
  const float* w_qkv = (const float*)d_in[1];
  const float* w_out = (const float*)d_in[2];
  const float* b_out = (const float*)d_in[3];
  float* out = (float*)d_out;

  char* ws = (char*)d_ws;
  const size_t seg = (size_t)BATCH * HEADS * NTOK * DH * sizeof(_Float16); // 3 MB
  _Float16* qb = (_Float16*)(ws);
  _Float16* kb = (_Float16*)(ws + seg);
  _Float16* vb = (_Float16*)(ws + 2 * seg);
  _Float16* ao = (_Float16*)(ws + 3 * seg);  // [B, N, 384] f16, 3 MB

  qkv_kernel<<<dim3(QKVC / 64, (BATCH * NTOK) / 128), 256, 0, stream>>>(
      x, w_qkv, qb, kb, vb);
  attn_kernel<<<dim3(NTOK / 128, BATCH * HEADS), 256, 0, stream>>>(
      qb, kb, vb, ao);
  proj_kernel<<<dim3(HID / 64, (BATCH * NTOK) / 128), 256, 0, stream>>>(
      ao, w_out, b_out, x, out);
}
